// DeformableTransformerEncoderLayer_39453569581668
// MI455X (gfx1250) — matrix operand order repaired and checked by reading, not compiled, not run on hardware
//
#include <hip/hip_runtime.h>
#include <hip/hip_bf16.h>
#include <math.h>

// ---------------------------------------------------------------------------
// Problem constants (from reference)
// ---------------------------------------------------------------------------
#define BSZ 2
#define DMODEL 256
#define NHEADS 8
#define NLVL 4
#define NPTS 4
#define WSZ 7
#define DFFN 1024
#define LENQ 11253
#define DHEAD 32
#define MROWS (BSZ * LENQ)            // 22506 flattened GEMM rows
#define KMAX 1024
#define LDS_PAD 8                     // u16 elems; stride%64banks==4 -> conflict-free

__device__ __constant__ int d_lvl_start[NLVL] = {0, 8464, 10580, 11109};
__device__ __constant__ int d_lvl_hw[NLVL]    = {92, 46, 23, 12};

typedef unsigned short u16;
typedef __attribute__((ext_vector_type(16))) __bf16 v16bf;
typedef __attribute__((ext_vector_type(8)))  float  v8f;

__device__ __forceinline__ u16 f32_to_bf16_rne(float f) {
    unsigned int u = __float_as_uint(f);
    unsigned int lsb = (u >> 16) & 1u;
    u += 0x7fffu + lsb;                       // round-to-nearest-even
    return (u16)(u >> 16);
}

// ---------------------------------------------------------------------------
// Conversion kernels: f32 -> bf16, and (K,N) f32 -> (N,K) bf16 transpose
// ---------------------------------------------------------------------------
__global__ void cvt_f32_bf16_kernel(const float* __restrict__ in,
                                    u16* __restrict__ out, long n) {
    long i = (long)blockIdx.x * blockDim.x + threadIdx.x;
    long stride = (long)gridDim.x * blockDim.x;
    for (; i < n; i += stride) out[i] = f32_to_bf16_rne(in[i]);
}

__global__ void transpose_cvt_bf16_kernel(const float* __restrict__ W,
                                          u16* __restrict__ Wt, int K, int N) {
    long i = (long)blockIdx.x * blockDim.x + threadIdx.x;
    long n = (long)K * N;
    if (i >= n) return;
    int k = (int)(i / N);
    int c = (int)(i % N);
    Wt[(long)c * K + k] = f32_to_bf16_rne(W[i]);
}

// ---------------------------------------------------------------------------
// Kernel 1: per-(b,h,q) 7x7 window gather from parse-mask, stable top-4,
// emit offsets (dx,dy) and softmax attention weights (with the 1/NL factor
// from the softmax over NL*NP broadcast values).
// ---------------------------------------------------------------------------
__global__ __launch_bounds__(256)
void topk_offsets_kernel(const float* __restrict__ m0, const float* __restrict__ m1,
                         const float* __restrict__ m2, const float* __restrict__ m3,
                         const float* __restrict__ pm_w, const float* __restrict__ pm_b,
                         float* __restrict__ offs,   // (BSZ,LENQ,NHEADS,NPTS,2)
                         float* __restrict__ attnw)  // (BSZ,LENQ,NHEADS,NPTS)
{
    int gid = blockIdx.x * blockDim.x + threadIdx.x;
    if (gid >= BSZ * NHEADS * LENQ) return;
    int b = gid / (NHEADS * LENQ);
    int h = (gid / LENQ) % NHEADS;
    int q = gid % LENQ;

    int l = 3;
    if (q < d_lvl_start[1]) l = 0;
    else if (q < d_lvl_start[2]) l = 1;
    else if (q < d_lvl_start[3]) l = 2;
    int W_ = d_lvl_hw[l];
    int qq = q - d_lvl_start[l];
    int r = qq / W_, c = qq % W_;

    const float* ml = (l == 0) ? m0 : (l == 1) ? m1 : (l == 2) ? m2 : m3;
    const float* mb = ml + (long)b * W_ * W_;
    float sw = pm_w[l * NHEADS + h];
    float sb = pm_b[l * NHEADS + h];

    float v0 = -3.0e38f, v1 = -3.0e38f, v2 = -3.0e38f, v3 = -3.0e38f;
    int i0 = 0, i1 = 0, i2 = 0, i3 = 0;
    #pragma unroll
    for (int i = 0; i < WSZ; ++i) {
        int yy = r + i - 3;
        #pragma unroll
        for (int j = 0; j < WSZ; ++j) {
            int xx = c + j - 3;
            int k = i * WSZ + j;
            float val = 0.0f;   // zero-padded (pad happens AFTER scale+bias)
            if (yy >= 0 && yy < W_ && xx >= 0 && xx < W_)
                val = mb[yy * W_ + xx] * sw + sb;
            if (val > v0)      { v3=v2;i3=i2; v2=v1;i2=i1; v1=v0;i1=i0; v0=val;i0=k; }
            else if (val > v1) { v3=v2;i3=i2; v2=v1;i2=i1; v1=val;i1=k; }
            else if (val > v2) { v3=v2;i3=i2; v2=val;i2=k; }
            else if (val > v3) { v3=val;i3=k; }
        }
    }
    // softmax over NL*NP broadcast copies: w_p = exp(v_p-max) / (NL * sum)
    float e0 = __expf(v0 - v0), e1 = __expf(v1 - v0);
    float e2 = __expf(v2 - v0), e3 = __expf(v3 - v0);
    float inv = 1.0f / ((float)NLVL * (e0 + e1 + e2 + e3));

    long base = (((long)b * LENQ + q) * NHEADS + h) * NPTS;
    float vs[4] = {e0 * inv, e1 * inv, e2 * inv, e3 * inv};
    int   is[4] = {i0, i1, i2, i3};
    #pragma unroll
    for (int p = 0; p < NPTS; ++p) {
        int k = is[p];
        offs[(base + p) * 2 + 0] = (float)(k % WSZ - 3);   // dx
        offs[(base + p) * 2 + 1] = (float)(k / WSZ - 3);   // dy
        attnw[base + p] = vs[p];
    }
}

// ---------------------------------------------------------------------------
// WMMA bf16 GEMM: C[M,N] = A[M,K] @ Bt[N,K]^T (+bias) [+relu] -> f32 or bf16.
//
// Block = 256 threads (8 waves) computes a 16 x 256 output tile:
//  * A tile (16 rows x K) is staged once into LDS cooperatively, with
//    out-of-range rows zero-filled at staging time (no per-iter cndmask).
//    Row stride K+8 u16 => lane-to-lane stride % 64 banks == 4 => the 16-lane
//    column reads of ds_load_b128 are bank-conflict free.
//  * Each wave owns TWO 16x16 N-tiles (cols w*16 and 128+w*16), so every A
//    fragment read from LDS feeds two v_wmma issues.
//
// Fragment layouts per CDNA5 ISA (wave32):
//  A (16x32 bf16): lanes 0-15 row=lane, halves = K {k0..k0+7, k0+16..k0+23};
//                  lanes 16-31 row=lane-16, K {k0+8..15, k0+24..31}.
//  B (32x16 bf16): lane col = lane&15; lanes 0-15 hold K {k0..k0+7, k0+16..k0+23},
//                  lanes 16-31 hold K {k0+8..15, k0+24..31} (same as A).
//  C/D (16x16 f32): lanes 0-15 col=lane rows 0..7; lanes 16-31 rows 8..15.
// EPI: 0 = f32 + bias, 1 = bf16 + bias + relu
// ---------------------------------------------------------------------------
template <int EPI>
__global__ __launch_bounds__(256)
void gemm_bf16_wmma_kernel(const u16* __restrict__ A, const u16* __restrict__ Bt,
                           const float* __restrict__ bias,
                           float* __restrict__ Cf, u16* __restrict__ Cb,
                           int M, int N, int K)
{
    __shared__ __align__(16) u16 As[16 * (KMAX + LDS_PAD)];

    const int wave = threadIdx.x >> 5;
    const int lane = threadIdx.x & 31;
    const int tileM = blockIdx.x * 16;
    const int tileN = blockIdx.y * 256 + wave * 16;     // first of two tiles
    const int rstride = K + LDS_PAD;
    const uint4 z4 = {0u, 0u, 0u, 0u};

    // ---- cooperative A-tile staging: global -> LDS (zero-fill OOB rows) ----
    const int kv = K >> 3;                              // uint4 chunks per row
    for (int c = threadIdx.x; c < 16 * kv; c += 256) {
        int r  = c / kv;
        int kc = (c - r * kv) << 3;
        uint4 d = z4;
        int row = tileM + r;
        if (row < M)
            d = *reinterpret_cast<const uint4*>(A + (long)row * K + kc);
        *reinterpret_cast<uint4*>(&As[r * rstride + kc]) = d;
    }
    __syncthreads();

    // ---- per-wave fragment pointers ----
    const int  half16 = lane & 15;
    const int  koffA  = (lane < 16) ? 0 : 8;
    const int  koffB  = (lane < 16) ? 0 : 8;
    const u16* Arow = &As[half16 * rstride + koffA];            // LDS
    const u16* Bp0  = Bt + (long)(tileN + half16) * K + koffB;  // global
    const u16* Bp1  = Bt + (long)(tileN + 128 + half16) * K + koffB;

    v8f acc0 = {0.f, 0.f, 0.f, 0.f, 0.f, 0.f, 0.f, 0.f};
    v8f acc1 = {0.f, 0.f, 0.f, 0.f, 0.f, 0.f, 0.f, 0.f};

    for (int k0 = 0; k0 < K; k0 += 32) {
        union { uint4 u[2]; v16bf v; } fa, fb0, fb1;
        fa.u[0]  = *reinterpret_cast<const uint4*>(Arow + k0);        // ds_load_b128
        fa.u[1]  = *reinterpret_cast<const uint4*>(Arow + k0 + 16);
        fb0.u[0] = *reinterpret_cast<const uint4*>(Bp0 + k0);
        fb0.u[1] = *reinterpret_cast<const uint4*>(Bp0 + k0 + 16);
        fb1.u[0] = *reinterpret_cast<const uint4*>(Bp1 + k0);
        fb1.u[1] = *reinterpret_cast<const uint4*>(Bp1 + k0 + 16);
        if (k0 + 32 < K) {   // gfx1250 global_prefetch_b8 path
            __builtin_prefetch(Bp0 + k0 + 32, 0, 1);
            __builtin_prefetch(Bp1 + k0 + 32, 0, 1);
        }
        acc0 = __builtin_amdgcn_wmma_f32_16x16x32_bf16(
            false, fa.v, false, fb0.v, (short)0, acc0, false, false);
        acc1 = __builtin_amdgcn_wmma_f32_16x16x32_bf16(
            false, fa.v, false, fb1.v, (short)0, acc1, false, false);
    }

    // ---- epilogue ----
    const int col0 = tileN + half16;
    const int col1 = col0 + 128;
    const float bc0 = bias ? bias[col0] : 0.0f;
    const float bc1 = bias ? bias[col1] : 0.0f;
    const int rbase = tileM + ((lane < 16) ? 0 : 8);
    #pragma unroll
    for (int r = 0; r < 8; ++r) {
        int row = rbase + r;
        if (row >= M) continue;
        float v0 = acc0[r] + bc0;
        float v1 = acc1[r] + bc1;
        if (EPI == 1) {
            v0 = fmaxf(v0, 0.0f);
            v1 = fmaxf(v1, 0.0f);
            Cb[(long)row * N + col0] = f32_to_bf16_rne(v0);
            Cb[(long)row * N + col1] = f32_to_bf16_rne(v1);
        } else {
            Cf[(long)row * N + col0] = v0;
            Cf[(long)row * N + col1] = v1;
        }
    }
}

// ---------------------------------------------------------------------------
// Kernel 3: multi-scale deformable attention aggregation.
// One wave per (b,q,h); lane = channel (DHEAD == warp size == 32).
// 4 levels x 4 points x 4 bilinear corners; each corner load is one
// coalesced 128B transaction (32 consecutive f32 channels), mostly L2 hits
// since `value` is only 23 MB (<< 192 MB L2).
// ---------------------------------------------------------------------------
__global__ __launch_bounds__(256)
void msdeform_agg_kernel(const float* __restrict__ value,   // (BSZ,LENQ,DMODEL)
                         const float* __restrict__ refpts,  // (BSZ,LENQ,NLVL,2)
                         const float* __restrict__ offs,    // (BSZ,LENQ,NHEADS,NPTS,2)
                         const float* __restrict__ attnw,   // (BSZ,LENQ,NHEADS,NPTS)
                         u16* __restrict__ attn_out)        // (BSZ,LENQ,DMODEL) bf16
{
    int wid = blockIdx.x * 8 + (threadIdx.x >> 5);
    int lane = threadIdx.x & 31;
    if (wid >= BSZ * LENQ * NHEADS) return;
    int b = wid / (LENQ * NHEADS);
    int rem = wid % (LENQ * NHEADS);
    int q = rem / NHEADS;
    int h = rem % NHEADS;

    const float* vb = value + ((long)b * LENQ) * DMODEL + h * DHEAD + lane;
    long obase = (((long)b * LENQ + q) * NHEADS + h) * NPTS;

    float acc = 0.0f;
    #pragma unroll
    for (int l = 0; l < NLVL; ++l) {
        int W_ = d_lvl_hw[l];
        int s0 = d_lvl_start[l];
        float rx = refpts[(((long)b * LENQ + q) * NLVL + l) * 2 + 0];
        float ry = refpts[(((long)b * LENQ + q) * NLVL + l) * 2 + 1];
        #pragma unroll
        for (int p = 0; p < NPTS; ++p) {
            float ox = offs[(obase + p) * 2 + 0];
            float oy = offs[(obase + p) * 2 + 1];
            float aw = attnw[obase + p];
            // (ref + off/W)*W - 0.5  ==  ref*W + off - 0.5
            float x = rx * (float)W_ + ox - 0.5f;
            float y = ry * (float)W_ + oy - 0.5f;
            float x0f = floorf(x), y0f = floorf(y);
            int x0 = (int)x0f, y0 = (int)y0f;
            float tx = x - x0f, ty = y - y0f;
            float w00 = (1.f - tx) * (1.f - ty) * aw;
            float w01 = tx * (1.f - ty) * aw;
            float w10 = (1.f - tx) * ty * aw;
            float w11 = tx * ty * aw;
            #pragma unroll
            for (int cr = 0; cr < 4; ++cr) {
                int yi = y0 + (cr >> 1);
                int xi = x0 + (cr & 1);
                float w = (cr == 0) ? w00 : (cr == 1) ? w01 : (cr == 2) ? w10 : w11;
                if (xi >= 0 && xi < W_ && yi >= 0 && yi < W_)
                    acc += w * vb[(long)(s0 + yi * W_ + xi) * DMODEL];
            }
        }
    }
    attn_out[((long)b * LENQ + q) * DMODEL + h * DHEAD + lane] = f32_to_bf16_rne(acc);
}

// ---------------------------------------------------------------------------
// Residual add + LayerNorm over D=256. One block (256 threads) per row.
// Optionally also emits a bf16 copy for the next WMMA GEMM.
// ---------------------------------------------------------------------------
__global__ __launch_bounds__(256)
void add_layernorm_kernel(const float* __restrict__ resid,  // (rows, 256)
                          const float* __restrict__ delta,  // (rows, 256)
                          const float* __restrict__ g, const float* __restrict__ bb,
                          float* __restrict__ xout, u16* __restrict__ xbf)
{
    __shared__ float s1[256];
    __shared__ float s2[256];
    long row = blockIdx.x;
    int i = threadIdx.x;
    float v = resid[row * DMODEL + i] + delta[row * DMODEL + i];
    s1[i] = v;
    s2[i] = v * v;
    __syncthreads();
    for (int st = 128; st > 0; st >>= 1) {
        if (i < st) { s1[i] += s1[i + st]; s2[i] += s2[i + st]; }
        __syncthreads();
    }
    float mu = s1[0] * (1.0f / DMODEL);
    float var = s2[0] * (1.0f / DMODEL) - mu * mu;
    float rs = rsqrtf(var + 1e-5f);
    float y = (v - mu) * rs * g[i] + bb[i];
    if (xout) xout[row * DMODEL + i] = y;
    if (xbf)  xbf[row * DMODEL + i] = f32_to_bf16_rne(y);
}

// ---------------------------------------------------------------------------
// Host-side launcher
// ---------------------------------------------------------------------------
extern "C" void kernel_launch(void* const* d_in, const int* in_sizes, int n_in,
                              void* d_out, int out_size, void* d_ws, size_t ws_size,
                              hipStream_t stream) {
    const float* src    = (const float*)d_in[0];
    // d_in[1] = pos (unused by reference)
    const float* refpts = (const float*)d_in[2];
    // d_in[3] = window_grid (computed analytically)
    const float* m0 = (const float*)d_in[4];
    const float* m1 = (const float*)d_in[5];
    const float* m2 = (const float*)d_in[6];
    const float* m3 = (const float*)d_in[7];
    // d_in[8] spatial_shapes, d_in[9] level_start_index (hardcoded)
    const float* pm_w = (const float*)d_in[10];
    const float* pm_b = (const float*)d_in[11];
    const float* Wv = (const float*)d_in[12];
    const float* bv = (const float*)d_in[13];
    const float* Wo = (const float*)d_in[14];
    const float* bo = (const float*)d_in[15];
    const float* ln1_g = (const float*)d_in[16];
    const float* ln1_b = (const float*)d_in[17];
    const float* W1 = (const float*)d_in[18];
    const float* b1 = (const float*)d_in[19];
    const float* W2 = (const float*)d_in[20];
    const float* b2 = (const float*)d_in[21];
    const float* ln2_g = (const float*)d_in[22];
    const float* ln2_b = (const float*)d_in[23];
    float* out = (float*)d_out;

    // ---- workspace layout (with lifetime-based buffer reuse) ----
    char* ws = (char*)d_ws;
    size_t cur = 0;
    auto alloc = [&](size_t bytes) -> void* {
        void* p = ws + cur;
        cur += (bytes + 255) & ~(size_t)255;
        return p;
    };
    const long MD  = (long)MROWS * DMODEL;     // 22506*256
    const long MF  = (long)MROWS * DFFN;       // 22506*1024
    float* offs    = (float*)alloc((long)BSZ * LENQ * NHEADS * NPTS * 2 * 4);
    float* attnw   = (float*)alloc((long)BSZ * LENQ * NHEADS * NPTS * 4);
    float* regA    = (float*)alloc(MD * 4);    // value -> proj -> ffn_out
    u16*   regB    = (u16*)  alloc(MD * 2);    // src_bf16 -> attn_out_bf16 -> x_bf16
    float* x_f32   = (float*)alloc(MD * 4);
    u16*   h_bf16  = (u16*)  alloc(MF * 2);
    u16*   WvT = (u16*)alloc((long)DMODEL * DMODEL * 2);
    u16*   WoT = (u16*)alloc((long)DMODEL * DMODEL * 2);
    u16*   W1T = (u16*)alloc((long)DMODEL * DFFN * 2);
    u16*   W2T = (u16*)alloc((long)DFFN * DMODEL * 2);
    (void)ws_size; (void)in_sizes; (void)n_in; (void)out_size;

    const int tilesM = (MROWS + 15) / 16;      // 1407

    // 0) weight transpose+convert, src convert
    transpose_cvt_bf16_kernel<<<(DMODEL * DMODEL + 255) / 256, 256, 0, stream>>>(Wv, WvT, DMODEL, DMODEL);
    transpose_cvt_bf16_kernel<<<(DMODEL * DMODEL + 255) / 256, 256, 0, stream>>>(Wo, WoT, DMODEL, DMODEL);
    transpose_cvt_bf16_kernel<<<(DMODEL * DFFN + 255) / 256, 256, 0, stream>>>(W1, W1T, DMODEL, DFFN);
    transpose_cvt_bf16_kernel<<<(DFFN * DMODEL + 255) / 256, 256, 0, stream>>>(W2, W2T, DFFN, DMODEL);
    cvt_f32_bf16_kernel<<<4096, 256, 0, stream>>>(src, regB, MD);

    // 1) mask top-k -> offsets + softmax weights
    topk_offsets_kernel<<<(BSZ * NHEADS * LENQ + 255) / 256, 256, 0, stream>>>(
        m0, m1, m2, m3, pm_w, pm_b, offs, attnw);

    // 2) value = src @ Wv + bv      (WMMA bf16; N=256 -> grid.y = 1)
    gemm_bf16_wmma_kernel<0><<<dim3(tilesM, DMODEL / 256), 256, 0, stream>>>(
        regB, WvT, bv, regA, nullptr, MROWS, DMODEL, DMODEL);

    // 3) deformable attention aggregation -> bf16 attn_out (reuses regB)
    msdeform_agg_kernel<<<(BSZ * LENQ * NHEADS) / 8, 256, 0, stream>>>(
        regA, refpts, offs, attnw, regB);

    // 4) proj = attn_out @ Wo + bo  (WMMA bf16) -> regA
    gemm_bf16_wmma_kernel<0><<<dim3(tilesM, DMODEL / 256), 256, 0, stream>>>(
        regB, WoT, bo, regA, nullptr, MROWS, DMODEL, DMODEL);

    // 5) x = LN1(src + proj); emit f32 + bf16 (bf16 reuses regB)
    add_layernorm_kernel<<<MROWS, 256, 0, stream>>>(src, regA, ln1_g, ln1_b, x_f32, regB);

    // 6) h = relu(x @ W1 + b1)      (WMMA bf16, bf16 out; N=1024 -> grid.y = 4)
    gemm_bf16_wmma_kernel<1><<<dim3(tilesM, DFFN / 256), 256, 0, stream>>>(
        regB, W1T, b1, nullptr, h_bf16, MROWS, DFFN, DMODEL);

    // 7) y = h @ W2 + b2            (WMMA bf16, K=1024) -> regA
    gemm_bf16_wmma_kernel<0><<<dim3(tilesM, DMODEL / 256), 256, 0, stream>>>(
        h_bf16, W2T, b2, regA, nullptr, MROWS, DMODEL, DFFN);

    // 8) out = LN2(x + y)
    add_layernorm_kernel<<<MROWS, 256, 0, stream>>>(x_f32, regA, ln2_g, ln2_b, out, nullptr);
}
